// SelfTransFormer_86526411145603
// MI455X (gfx1250) — matrix operand order repaired and checked by reading, not compiled, not run on hardware
//
#include <hip/hip_runtime.h>
#include <math.h>

#define B_    8
#define CIN_  512
#define COUT_ 1024
#define P_    2048
#define H_    8
#define DH_   128
#define EPS_  1e-5f
#define SLOPE_ 0.1f
#define RSQ_DH 0.08838834764831845f   // 1/sqrt(128)

typedef __bf16 v16bf __attribute__((ext_vector_type(16)));
typedef __bf16 v2bf  __attribute__((ext_vector_type(2)));
typedef float  v2f   __attribute__((ext_vector_type(2)));
typedef float  v8f   __attribute__((ext_vector_type(8)));
typedef int    v4i   __attribute__((vector_size(16)));   // matches builtin param

#define AS1 __attribute__((address_space(1)))
#define AS3 __attribute__((address_space(3)))

#if defined(__gfx1250__) && __has_builtin(__builtin_amdgcn_global_load_async_to_lds_b128)
#define HAVE_ASYNC_LDS 1
#else
#define HAVE_ASYNC_LDS 0
#endif

__device__ __forceinline__ void async_wait0() {
#if HAVE_ASYNC_LDS
#if __has_builtin(__builtin_amdgcn_s_wait_asynccnt)
  __builtin_amdgcn_s_wait_asynccnt(0);
#else
  asm volatile("s_wait_asynccnt 0x0" ::: "memory");
#endif
#endif
}

union Frag16 { v16bf v; uint4 q[2]; unsigned short h[16]; };
union H8 { uint4 q; unsigned int w[4]; unsigned short h[8]; };

// pack two f32 -> bf16x2; __builtin_convertvector lets ISel pick
// v_cvt_pk_bf16_f32 when the target has it.
__device__ __forceinline__ unsigned int f2bf2(float lo, float hi) {
  v2f f = {lo, hi};
  union { v2bf v; unsigned int u; } cv;
  cv.v = __builtin_convertvector(f, v2bf);
  return cv.u;
}
__device__ __forceinline__ unsigned short f2bf(float f) {
  return (unsigned short)(f2bf2(f, 0.f) & 0xFFFFu);
}
__device__ __forceinline__ float bf2f(unsigned short h) {
  return __uint_as_float(((unsigned int)h) << 16);
}

// ---------------------------------------------------------------------------
// Kernel 1: Y[mat,b] = W_mat x_b  (bf16 WMMA, f32 accum), write Y as bf16,
// accumulate per-channel sum / sumsq from the f32 accumulators.
// Block: 256 threads (8 waves), tile 64(out-ch) x 64(pos), K-loop over 512.
// ---------------------------------------------------------------------------
__global__ __launch_bounds__(256) void proj_gemm_bnstat(
    const float* __restrict__ x,
    const float* __restrict__ Wq, const float* __restrict__ Wk,
    const float* __restrict__ Wv,
    unsigned short* __restrict__ Y, float* __restrict__ stats)
{
  const int t = threadIdx.x;
  const int wave = t >> 5, lane = t & 31;
  const int laneM = lane & 15, hi = lane >> 4;
  const int z = blockIdx.z;
  const int mat = z / B_, b = z % B_;
  const float* W = (mat == 0) ? Wq : ((mat == 1) ? Wk : Wv);
  const int m0 = blockIdx.y * 64;   // out-channel base
  const int n0 = blockIdx.x * 64;   // position base

  __shared__ alignas(16) unsigned short As[64 * 32];  // [m][k] bf16
  __shared__ alignas(16) unsigned short Bs[64 * 32];  // [n][k] bf16 (x transposed)
  __shared__ float csum[64], csq[64];
  if (t < 64) { csum[t] = 0.f; csq[t] = 0.f; }

  const int tr = wave >> 1;        // C-tile row 0..3
  const int tc = (wave & 1) * 2;   // C-tile col base 0 or 2
  v8f acc0 = {0.f,0.f,0.f,0.f,0.f,0.f,0.f,0.f};
  v8f acc1 = {0.f,0.f,0.f,0.f,0.f,0.f,0.f,0.f};

  // Bs staging geometry (fixed per thread): c-pair x 4-wide n group
  const int kp = t >> 4;           // 0..15  -> kk = 2*kp
  const int ng = (t & 15) * 4;     // n base 0..60

  for (int k0 = 0; k0 < CIN_; k0 += 32) {
    __syncthreads();
    // ---- A tile: row-major, vectorized float4 -> packed bf16x4 ----
#pragma unroll
    for (int i = 0; i < 2; ++i) {
      int cid = t + i * 256;                  // 512 chunks of 4 halfs
      int base = cid * 4;
      int m = base >> 5, k = base & 31;
      float4 w = *(const float4*)(W + (size_t)(m0 + m) * CIN_ + k0 + k);
      uint2 pk;
      pk.x = f2bf2(w.x, w.y);
      pk.y = f2bf2(w.z, w.w);
      *(uint2*)(As + base) = pk;
    }
    // ---- B tile: transpose x[k][n] -> Bs[n][k], pack k-pairs as b32 ----
    {
      int kk = kp * 2;
      const float* xr = x + ((size_t)b * CIN_ + k0 + kk) * P_ + n0 + ng;
      float4 xa = *(const float4*)(xr);
      float4 xb = *(const float4*)(xr + P_);
      *(unsigned int*)(Bs + (ng + 0) * 32 + kk) = f2bf2(xa.x, xb.x);
      *(unsigned int*)(Bs + (ng + 1) * 32 + kk) = f2bf2(xa.y, xb.y);
      *(unsigned int*)(Bs + (ng + 2) * 32 + kk) = f2bf2(xa.z, xb.z);
      *(unsigned int*)(Bs + (ng + 3) * 32 + kk) = f2bf2(xa.w, xb.w);
    }
    __syncthreads();
    Frag16 a, b0f, b1f;
    const unsigned short* arow = As + (tr * 16 + laneM) * 32;
    a.q[0] = *(const uint4*)(arow + (hi ? 8 : 0));
    a.q[1] = *(const uint4*)(arow + (hi ? 24 : 16));
    const unsigned short* brow0 = Bs + (tc * 16 + laneM) * 32;
    b0f.q[0] = *(const uint4*)(brow0 + (hi ? 8 : 0));
    b0f.q[1] = *(const uint4*)(brow0 + (hi ? 24 : 16));
    const unsigned short* brow1 = Bs + ((tc + 1) * 16 + laneM) * 32;
    b1f.q[0] = *(const uint4*)(brow1 + (hi ? 8 : 0));
    b1f.q[1] = *(const uint4*)(brow1 + (hi ? 24 : 16));
    acc0 = __builtin_amdgcn_wmma_f32_16x16x32_bf16(false, a.v, false, b0f.v,
                                                   (short)0, acc0, false, false);
    acc1 = __builtin_amdgcn_wmma_f32_16x16x32_bf16(false, a.v, false, b1f.v,
                                                   (short)0, acc1, false, false);
  }

  unsigned short* Yp = Y + ((size_t)mat * B_ + b) * COUT_ * P_;
#pragma unroll
  for (int r = 0; r < 8; ++r) {
    int lo = tr * 16 + r + hi * 8;                 // local channel in [0,64)
    int lp0 = tc * 16 + laneM, lp1 = (tc + 1) * 16 + laneM;
    float v0 = acc0[r], v1 = acc1[r];
    unsigned int pk = f2bf2(v0, v1);               // one packed cvt
    Yp[(size_t)(m0 + lo) * P_ + n0 + lp0] = (unsigned short)(pk & 0xFFFFu);
    Yp[(size_t)(m0 + lo) * P_ + n0 + lp1] = (unsigned short)(pk >> 16);
    atomicAdd(&csum[lo], v0 + v1);
    atomicAdd(&csq[lo], v0 * v0 + v1 * v1);
  }
  __syncthreads();
  if (t < 64) {
    atomicAdd(&stats[(size_t)(mat * COUT_ + m0 + t) * 2 + 0], csum[t]);
    atomicAdd(&stats[(size_t)(mat * COUT_ + m0 + t) * 2 + 1], csq[t]);
  }
}

// ---------------------------------------------------------------------------
// Kernel 2: per-channel BN scale/shift from accumulated stats.
// ---------------------------------------------------------------------------
__global__ void bn_finalize(const float* __restrict__ stats,
                            const float* __restrict__ gq, const float* __restrict__ bq,
                            const float* __restrict__ gk, const float* __restrict__ bk,
                            const float* __restrict__ gv, const float* __restrict__ bv,
                            float* __restrict__ ss)
{
  int idx = blockIdx.x * blockDim.x + threadIdx.x;
  if (idx >= 3 * COUT_) return;
  int mat = idx / COUT_, o = idx % COUT_;
  const float* g  = (mat == 0) ? gq : ((mat == 1) ? gk : gv);
  const float* be = (mat == 0) ? bq : ((mat == 1) ? bk : bv);
  const float cnt = (float)(B_ * P_);
  float s = stats[idx * 2], sq = stats[idx * 2 + 1];
  float mean = s / cnt;
  float var  = sq / cnt - mean * mean;
  float sc = g[o] * rsqrtf(var + EPS_);
  ss[idx * 2 + 0] = sc;
  ss[idx * 2 + 1] = be[o] - mean * sc;
}

// ---------------------------------------------------------------------------
// Kernel 3: in-place BN apply + LeakyReLU(0.1) on bf16 Y. 8 elems/thread
// through uint4, pair-wise packed bf16 conversion.
// ---------------------------------------------------------------------------
__global__ __launch_bounds__(256) void bn_apply_lrelu(unsigned short* __restrict__ Y,
                                                      const float* __restrict__ ss)
{
  size_t i0 = ((size_t)blockIdx.x * 256 + threadIdx.x) * 8;
  int ch  = (int)((i0 / P_) % COUT_);
  int mat = (int)(i0 / ((size_t)B_ * COUT_ * P_));
  float sc = ss[(mat * COUT_ + ch) * 2 + 0];
  float sh = ss[(mat * COUT_ + ch) * 2 + 1];
  H8 v; v.q = *(const uint4*)(Y + i0);
#pragma unroll
  for (int k = 0; k < 4; ++k) {
    float f0 = bf2f(v.h[2 * k + 0]) * sc + sh;
    float f1 = bf2f(v.h[2 * k + 1]) * sc + sh;
    f0 = (f0 >= 0.f) ? f0 : SLOPE_ * f0;
    f1 = (f1 >= 0.f) ? f1 : SLOPE_ * f1;
    v.w[k] = f2bf2(f0, f1);
  }
  *(uint4*)(Y + i0) = v.q;
}

// ---------------------------------------------------------------------------
// Kernel 4: fused flash-style attention.  One block per (b, h, 16-row i-tile).
// S-tile = K^T V / sqrt(dh) via WMMA; online softmax; O += P Q^T via WMMA.
// Q chunk staged to LDS with GLOBAL_LOAD_ASYNC_TO_LDS_B128 (ASYNCcnt path),
// overlapped with the manual V transpose staging.
// ---------------------------------------------------------------------------
__global__ __launch_bounds__(256) void attn_kernel(const unsigned short* __restrict__ Y,
                                                   float* __restrict__ out)
{
  const int t = threadIdx.x;
  const int wave = t >> 5, lane = t & 31;
  const int laneM = lane & 15, hi = lane >> 4;
  const int b = blockIdx.z, h = blockIdx.y;
  const int i0 = blockIdx.x * 16;
  const int hc = h * DH_;

  const unsigned short* Q = Y + ((size_t)0 * B_ + b) * COUT_ * P_;
  const unsigned short* K = Y + ((size_t)1 * B_ + b) * COUT_ * P_;
  const unsigned short* V = Y + ((size_t)2 * B_ + b) * COUT_ * P_;

  __shared__ alignas(16) unsigned short Kt[16 * 128];  // [i][c]  (K^T tile)
  __shared__ alignas(16) unsigned short Vt[64 * 128];  // [j][c]  (V^T chunk)
  __shared__ alignas(16) unsigned short Qs[128 * 64];  // [c][j]  (Q chunk)
  __shared__ float Sbuf[16 * 64];
  __shared__ alignas(16) unsigned short Pbuf[16 * 64];
  __shared__ float m_row[16], l_row[16], corr_row[16];

  if (t < 16) { m_row[t] = -1e30f; l_row[t] = 0.f; }
  // K^T tile: one B128 load (8 halfs along i) -> 8 scattered b16 stores
  {
    int c = t >> 1, mg = (t & 1) * 8;
    H8 u; u.q = *(const uint4*)(K + (size_t)(hc + c) * P_ + i0 + mg);
#pragma unroll
    for (int e = 0; e < 8; ++e) Kt[(mg + e) * 128 + c] = u.h[e];
  }

  v8f acc = {0.f,0.f,0.f,0.f,0.f,0.f,0.f,0.f};
  const int cb = wave * 16;   // this wave's output-channel block

  for (int jt = 0; jt < P_; jt += 64) {
    __syncthreads();
    // ---- issue async DMA of Q chunk [128c x 64j] (transpose-free) ----
#pragma unroll
    for (int i = 0; i < 4; ++i) {
      int cid = t + i * 256;                  // 1024 chunks of 8 halfs
      int c = cid >> 3, nb = (cid & 7) * 8;
      const unsigned short* g = Q + (size_t)(hc + c) * P_ + jt + nb;
      unsigned short* l = Qs + c * 64 + nb;
#if HAVE_ASYNC_LDS
      __builtin_amdgcn_global_load_async_to_lds_b128((AS1 v4i*)g, (AS3 v4i*)l,
                                                     0, 0);
#else
      *(uint4*)l = *(const uint4*)g;
#endif
    }
    // ---- V transpose staging: c-pairs packed into b32 LDS stores ----
#pragma unroll
    for (int i = 0; i < 2; ++i) {
      int cid = t + i * 256;                  // 512 chunks
      int c = (cid >> 3) * 2, nb = (cid & 7) * 8;
      H8 ua, ub;
      ua.q = *(const uint4*)(V + (size_t)(hc + c) * P_ + jt + nb);
      ub.q = *(const uint4*)(V + (size_t)(hc + c + 1) * P_ + jt + nb);
#pragma unroll
      for (int e = 0; e < 8; ++e)
        *(unsigned int*)(Vt + (nb + e) * 128 + c) =
            (unsigned int)ua.h[e] | ((unsigned int)ub.h[e] << 16);
    }
    async_wait0();
    __syncthreads();

    if (wave < 4) {  // waves 0..3 compute the 4 S sub-tiles
      v8f s = {0.f,0.f,0.f,0.f,0.f,0.f,0.f,0.f};
#pragma unroll
      for (int cc = 0; cc < 128; cc += 32) {
        Frag16 a, bf;
        const unsigned short* ar = Kt + laneM * 128 + cc;
        a.q[0] = *(const uint4*)(ar + (hi ? 8 : 0));
        a.q[1] = *(const uint4*)(ar + (hi ? 24 : 16));
        const unsigned short* br = Vt + (wave * 16 + laneM) * 128 + cc;
        bf.q[0] = *(const uint4*)(br + (hi ? 8 : 0));
        bf.q[1] = *(const uint4*)(br + (hi ? 24 : 16));
        s = __builtin_amdgcn_wmma_f32_16x16x32_bf16(false, a.v, false, bf.v,
                                                    (short)0, s, false, false);
      }
#pragma unroll
      for (int r = 0; r < 8; ++r)
        Sbuf[(r + hi * 8) * 64 + wave * 16 + laneM] = s[r] * RSQ_DH;
    }
    __syncthreads();

    if (t < 16) {  // online softmax: one thread per row, packed P writes
      float mprev = m_row[t], mx = mprev;
      for (int j = 0; j < 64; ++j) mx = fmaxf(mx, Sbuf[t * 64 + j]);
      float corr = __expf(mprev - mx);
      float sum = l_row[t] * corr;
      for (int j = 0; j < 64; j += 2) {
        float e0 = __expf(Sbuf[t * 64 + j + 0] - mx);
        float e1 = __expf(Sbuf[t * 64 + j + 1] - mx);
        *(unsigned int*)(Pbuf + t * 64 + j) = f2bf2(e0, e1);
        sum += e0 + e1;
      }
      m_row[t] = mx; l_row[t] = sum; corr_row[t] = corr;
    }
    __syncthreads();

    // rescale O accumulator per row, then O += P * Q^T
#pragma unroll
    for (int r = 0; r < 8; ++r) acc[r] *= corr_row[r + hi * 8];
#pragma unroll
    for (int jc = 0; jc < 64; jc += 32) {
      Frag16 a, bf;
      const unsigned short* ar = Pbuf + laneM * 64 + jc;
      a.q[0] = *(const uint4*)(ar + (hi ? 8 : 0));
      a.q[1] = *(const uint4*)(ar + (hi ? 24 : 16));
      // B[k=j][n=c] = Q[c][jt+jc+k]: two 8-half runs (table k order) from the LDS Q chunk
      const unsigned short* br = Qs + (cb + laneM) * 64 + jc;
      bf.q[0] = *(const uint4*)(br + (hi ? 8 : 0));
      bf.q[1] = *(const uint4*)(br + (hi ? 24 : 16));
      acc = __builtin_amdgcn_wmma_f32_16x16x32_bf16(false, a.v, false, bf.v,
                                                    (short)0, acc, false, false);
    }
    if (jt + 64 < P_)  // hint: pull next V chunk toward the caches
      __builtin_prefetch(V + (size_t)hc * P_ + jt + 64 + (t & 63), 0, 0);
  }
  __syncthreads();

#pragma unroll
  for (int r = 0; r < 8; ++r) {
    int m = r + hi * 8;
    float val = acc[r] / l_row[m];
    out[((size_t)b * COUT_ + hc + cb + laneM) * P_ + i0 + m] = val;
  }
}

// ---------------------------------------------------------------------------
extern "C" void kernel_launch(void* const* d_in, const int* in_sizes, int n_in,
                              void* d_out, int out_size, void* d_ws, size_t ws_size,
                              hipStream_t stream) {
  (void)in_sizes; (void)n_in; (void)out_size; (void)ws_size;
  const float* x  = (const float*)d_in[0];
  const float* Wq = (const float*)d_in[1];
  const float* gq = (const float*)d_in[2];
  const float* bq = (const float*)d_in[3];
  const float* Wk = (const float*)d_in[4];
  const float* gk = (const float*)d_in[5];
  const float* bk = (const float*)d_in[6];
  const float* Wv = (const float*)d_in[7];
  const float* gv = (const float*)d_in[8];
  const float* bv = (const float*)d_in[9];
  float* out = (float*)d_out;

  const size_t yElems = (size_t)3 * B_ * COUT_ * P_;        // 50.3M bf16
  unsigned short* Ybf = (unsigned short*)d_ws;
  float* stats = (float*)((char*)d_ws + yElems * sizeof(unsigned short));
  float* ss = stats + 3 * COUT_ * 2;

  (void)hipMemsetAsync(stats, 0, (size_t)3 * COUT_ * 2 * sizeof(float), stream);

  dim3 gProj(P_ / 64, COUT_ / 64, 3 * B_);
  proj_gemm_bnstat<<<gProj, 256, 0, stream>>>(x, Wq, Wk, Wv, Ybf, stats);

  bn_finalize<<<(3 * COUT_ + 255) / 256, 256, 0, stream>>>(stats, gq, bq, gk, bk,
                                                           gv, bv, ss);

  bn_apply_lrelu<<<(unsigned)(yElems / 8 / 256), 256, 0, stream>>>(Ybf, ss);

  dim3 gAttn(P_ / 16, H_, B_);
  attn_kernel<<<gAttn, 256, 0, stream>>>(Ybf, out);   // 256 threads = 8 waves
}
